// MultiHeadLatentAttention_46359876993615
// MI455X (gfx1250) — matrix operand order repaired and checked by reading, not compiled, not run on hardware
//
#include <hip/hip_runtime.h>
#include <hip/hip_bf16.h>

// ---------------------------------------------------------------------------
// MLA forward for MI455X (gfx1250, wave32, WMMA bf16 16x16x32, fp32 accum)
// v2.1: 128x128x64 double-buffered GEMM (16 WMMA/wave/iter), TDM tile loads
//       (6-arg clang-23 builtin), flash attention with 64-key blocks.
// ---------------------------------------------------------------------------

typedef unsigned int u32;
typedef unsigned short u16;
typedef unsigned long long u64;
typedef __attribute__((ext_vector_type(16))) __bf16 v16bf;
typedef __attribute__((ext_vector_type(8)))  float  v8f;
typedef __attribute__((ext_vector_type(4)))  u32    v4u;
typedef __attribute__((ext_vector_type(8)))  int    v8i;
typedef __attribute__((ext_vector_type(4)))  int    v4i;

#define H_    2048
#define NH_   16
#define HD_   128
#define RD_   64
#define LAT_  256
#define B_    2
#define S_    2048
#define MTOK  (B_ * S_)   // 4096 token rows

#ifndef DISABLE_TDM
#if defined(__has_builtin)
#if __has_builtin(__builtin_amdgcn_tensor_load_to_lds)
#define HAVE_TDM 1
#endif
#endif
#endif
#ifndef HAVE_TDM
#define HAVE_TDM 0
#endif

// ----- bf16 helpers (round-to-nearest-even) --------------------------------
__device__ __forceinline__ u16 f2bf(float f) {
    u32 u = __float_as_uint(f);
    u32 r = u + 0x7fffu + ((u >> 16) & 1u);
    return (u16)(r >> 16);
}
__device__ __forceinline__ float bf2f(u16 b) {
    return __uint_as_float(((u32)b) << 16);
}

union FragU { v4u q[2]; v16bf v; };

__device__ __forceinline__ v16bf ld_frag2(const u16* p0, const u16* p1) {
    FragU f;
    f.q[0] = *(const v4u*)p0;
    f.q[1] = *(const v4u*)p1;
    return f.v;
}
__device__ __forceinline__ v16bf ld_frag(const u16* p) { return ld_frag2(p, p + 8); }

__device__ __forceinline__ v8f wmma_bf16(v16bf a, v16bf b, v8f c) {
    return __builtin_amdgcn_wmma_f32_16x16x32_bf16(false, a, false, b, (short)0, c,
                                                   false, false);
}

#if HAVE_TDM
// ---------------------------------------------------------------------------
// Tensor Data Mover: 2D tile (tileD0 elems x tileD1 rows, 2B elements) from
// global (row stride strideElems) into LDS at ldsByteAddr, with LDS padding
// of (padAmountCode+1) dwords after every 2^(padIntervalCode+1) dwords so the
// DMA directly produces our bank-conflict-free padded row strides.
// clang-23 6-arg builtin: (v4u g0, v8i g1, v4i g2, v4i g3, v8i unused, i32 cpol)
// ---------------------------------------------------------------------------
__device__ __forceinline__ void tdm_load_2d(u32 ldsByteAddr, const u16* gptr,
                                            u32 tileD0, u32 tileD1,
                                            u32 strideElems,
                                            u32 padIntervalCode, u32 padAmountCode) {
    u64 ga = (u64)(uintptr_t)gptr;
    v4u g0;
    g0[0] = 1u;                                           // count=1, user desc
    g0[1] = ldsByteAddr;                                  // lds_addr (bytes)
    g0[2] = (u32)ga;                                      // global_addr lo
    g0[3] = (u32)((ga >> 32) & 0x1ffffffu) | (2u << 30);  // global_addr hi | type=2
    v8i g1;
    g1[0] = (int)((1u << 16) |                            // data_size = 2B
                  (1u << 20) |                            // pad_enable
                  (padIntervalCode << 22) | (padAmountCode << 25));
    g1[1] = (int)((tileD0 & 0xffffu) << 16);              // tensor_dim0 lo16
    g1[2] = (int)(((tileD0 >> 16) & 0xffffu) | ((tileD1 & 0xffffu) << 16));
    g1[3] = (int)(((tileD1 >> 16) & 0xffffu) | ((tileD0 & 0xffffu) << 16)); // tile_dim0
    g1[4] = (int)(tileD1 & 0xffffu);                      // tile_dim1, tile_dim2=0
    g1[5] = (int)strideElems;                             // tensor_dim0_stride lo32
    g1[6] = 0;                                            // stride0 hi | stride1 lo
    g1[7] = 0;
    v4i g2; g2[0] = 1; g2[1] = 0; g2[2] = 0; g2[3] = 0;   // tensor_dim2=1
    v4i g3; g3[0] = 0; g3[1] = 0; g3[2] = 0; g3[3] = 0;
    v8i g4; g4[0] = 0; g4[1] = 0; g4[2] = 0; g4[3] = 0;
            g4[4] = 0; g4[5] = 0; g4[6] = 0; g4[7] = 0;   // unused 5th group
    __builtin_amdgcn_tensor_load_to_lds(g0, g1, g2, g3, g4, 0);
}
#endif

// ---------------------------------------------------------------------------
// fp32 -> bf16 conversion, 4 elements/thread
// ---------------------------------------------------------------------------
__global__ __launch_bounds__(256) void cvt_f32_bf16(const float* __restrict__ in,
                                                    u16* __restrict__ out, int n) {
    int i = (blockIdx.x * 256 + threadIdx.x) * 4;
    if (i >= n) return;
    float4 v = *(const float4*)(in + i);
    u32 lo = (u32)f2bf(v.x) | ((u32)f2bf(v.y) << 16);
    u32 hi = (u32)f2bf(v.z) | ((u32)f2bf(v.w) << 16);
    uint2 pk; pk.x = lo; pk.y = hi;
    *(uint2*)(out + i) = pk;
}

// ---------------------------------------------------------------------------
// GEMM: C[M,N] = A[M,K] @ B[N,K]^T  (A,B bf16, fp32 accum)
// Block tile 128x128, k-step 64, double-buffered LDS.
// 8 waves in 4(M) x 2(N): each wave 32x64 -> 16 WMMA per k-step.
// remap==1 scatters col n -> (n/64)*128 + (n%64) + colBase  (head packing)
// ---------------------------------------------------------------------------
#define BM 128
#define BN 128
#define BK 64
#define ASTR 72   // padded LDS row stride (bf16): 64 data + 8 pad, 144B (16B mult)

__global__ __launch_bounds__(256) void gemm_bf16_wmma(
    const u16* __restrict__ A, const u16* __restrict__ Bm,
    void* __restrict__ Cout, int K, int ldc, int remap, int colBase, int outF32)
{
    __shared__ u16 sA[2][BM * ASTR];
    __shared__ u16 sB[2][BN * ASTR];

    const int tid  = threadIdx.x;
    const int lane = tid & 31;
    const int wave = tid >> 5;
    const int l16  = lane & 15;
    const int hs   = lane >> 4;
    const int bm   = blockIdx.y * BM;
    const int bn   = blockIdx.x * BN;
    const int rm0  = (wave & 3) * 32;    // wave M origin in tile
    const int cn0  = (wave >> 2) * 64;   // wave N origin in tile

    v8f acc[2][4] = {};
    const int niter = K / BK;

#if HAVE_TDM
    if (wave == 0) {
        tdm_load_2d((u32)(uintptr_t)&sA[0][0], A + (size_t)bm * K, BK, BM, K, 4, 3);
        tdm_load_2d((u32)(uintptr_t)&sB[0][0], Bm + (size_t)bn * K, BK, BN, K, 4, 3);
    }
#else
    const int arow = tid >> 1, aseg = (tid & 1) * 32;     // 128 rows x 2 halves
    const u16* gA = A  + (size_t)(bm + arow) * K + aseg;
    const u16* gB = Bm + (size_t)(bn + arow) * K + aseg;
    v4u ra[4], rb[4];
#pragma unroll
    for (int j = 0; j < 4; ++j) {
        ra[j] = *(const v4u*)(gA + j * 8);
        rb[j] = *(const v4u*)(gB + j * 8);
    }
#pragma unroll
    for (int j = 0; j < 4; ++j) {
        *(v4u*)(&sA[0][0] + arow * ASTR + aseg + j * 8) = ra[j];
        *(v4u*)(&sB[0][0] + arow * ASTR + aseg + j * 8) = rb[j];
    }
    __syncthreads();
#endif

    for (int it = 0; it < niter; ++it) {
        const int buf = it & 1;
#if HAVE_TDM
        if (wave == 0) __builtin_amdgcn_s_wait_tensorcnt(0);
        __syncthreads();
        if (it + 1 < niter && wave == 0) {
            const int k1 = (it + 1) * BK;
            tdm_load_2d((u32)(uintptr_t)&sA[buf ^ 1][0], A + (size_t)bm * K + k1,
                        BK, BM, K, 4, 3);
            tdm_load_2d((u32)(uintptr_t)&sB[buf ^ 1][0], Bm + (size_t)bn * K + k1,
                        BK, BN, K, 4, 3);
        }
#else
        if (it + 1 < niter) {
            const int k1 = (it + 1) * BK;
#pragma unroll
            for (int j = 0; j < 4; ++j) {
                ra[j] = *(const v4u*)(gA + k1 + j * 8);
                rb[j] = *(const v4u*)(gB + k1 + j * 8);
            }
            __builtin_prefetch(gA + k1 + BK, 0, 3);
            __builtin_prefetch(gB + k1 + BK, 0, 3);
        }
#endif
        // ---- compute 32x64 from LDS buf: 2 k-halves x (2 A x 4 B) WMMAs ----
        const u16* sAb = &sA[buf][0];
        const u16* sBb = &sB[buf][0];
#pragma unroll
        for (int kh = 0; kh < 2; ++kh) {
            const int ko = kh * 32;
            const u16* pa0 = sAb + (rm0 + l16) * ASTR + ko + hs * 8;
            const u16* pa1 = pa0 + 16 * ASTR;
            v16bf a0 = ld_frag2(pa0, pa0 + 16);
            v16bf a1 = ld_frag2(pa1, pa1 + 16);
            const u16* pb = sBb + (cn0 + l16) * ASTR + ko + hs * 8;
#pragma unroll
            for (int nt = 0; nt < 4; ++nt) {
                v16bf bf = ld_frag2(pb + nt * 16 * ASTR, pb + nt * 16 * ASTR + 16);
                acc[0][nt] = wmma_bf16(a0, bf, acc[0][nt]);
                acc[1][nt] = wmma_bf16(a1, bf, acc[1][nt]);
            }
        }
#if HAVE_TDM
        __syncthreads();
#else
        if (it + 1 < niter) {
#pragma unroll
            for (int j = 0; j < 4; ++j) {
                *(v4u*)(&sA[buf ^ 1][0] + arow * ASTR + aseg + j * 8) = ra[j];
                *(v4u*)(&sB[buf ^ 1][0] + arow * ASTR + aseg + j * 8) = rb[j];
            }
        }
        __syncthreads();
#endif
    }

    // ---- epilogue ----
#pragma unroll
    for (int mi = 0; mi < 2; ++mi) {
#pragma unroll
        for (int nt = 0; nt < 4; ++nt) {
            int n  = bn + cn0 + nt * 16 + l16;
            int oc = (remap ? (((n >> 6) << 7) + (n & 63)) : n) + colBase;
#pragma unroll
            for (int r = 0; r < 8; ++r) {
                int m = bm + rm0 + mi * 16 + r + 8 * hs;
                float val = acc[mi][nt][r];
                if (outF32) ((float*)Cout)[(size_t)m * ldc + oc] = val;
                else        ((u16*)Cout)[(size_t)m * ldc + oc]   = f2bf(val);
            }
        }
    }
}

// ---------------------------------------------------------------------------
// RoPE in-place on packed (tok, h*128 + d) buffer, rotating d in [64,128)
// ---------------------------------------------------------------------------
__global__ __launch_bounds__(256) void rope_kernel(u16* __restrict__ buf) {
    int idx = blockIdx.x * 256 + threadIdx.x;
    int i   = idx & 31;
    int h   = (idx >> 5) & (NH_ - 1);
    int tok = idx >> 9;
    int s   = tok & (S_ - 1);
    float ang = (float)s * __expf(-(float)i * (9.21034037197618f / 32.0f));
    float sn, cs;
    __sincosf(ang, &sn, &cs);
    size_t base = (size_t)tok * H_ + h * HD_ + RD_;
    float x1 = bf2f(buf[base + i]);
    float x2 = bf2f(buf[base + 32 + i]);
    buf[base + i]      = f2bf(x1 * cs - x2 * sn);
    buf[base + 32 + i] = f2bf(x1 * sn + x2 * cs);
}

// ---------------------------------------------------------------------------
// Flash attention (causal), head dim 128, bf16 in/out, fp32 online softmax.
// Grid: (S/128, NH, B). 8 waves; wave owns 16 query rows.
// Key blocks of 64: 16 QK^T WMMAs + 16 PV WMMAs per wave per block.
// K staged by TDM (or manual), V transposed manually, P via wave-private LDS.
// ---------------------------------------------------------------------------
#define QB   128
#define KBLK 64
#define KSTR 136   // 128 data + 8 pad elems
#define VSTR 72    // 64 keys + 8 pad
#define PSTR 72

__global__ __launch_bounds__(256) void mla_flash(
    const u16* __restrict__ Q, const u16* __restrict__ Kk,
    const u16* __restrict__ V, u16* __restrict__ Y)
{
    __shared__ u16 sK[KBLK * KSTR];
    __shared__ u16 sV[HD_ * VSTR];       // transposed: [dv][key]
    __shared__ u16 sP[8 * 16 * PSTR];    // per-wave P tile (16 x 64)

    const int tid  = threadIdx.x;
    const int lane = tid & 31;
    const int wave = tid >> 5;
    const int l16  = lane & 15;
    const int hs   = lane >> 4;
    const int q0   = blockIdx.x * QB;
    const int h    = blockIdx.y;
    const int b    = blockIdx.z;
    const size_t rowBase = (size_t)b * S_;
    const int hcol = h * HD_;
    const float scale = 0.08838834764831845f;   // 1/sqrt(128)

    // Q fragments for this wave's 16 rows: 4 chunks of 32 along d
    v16bf qf[4];
    {
        const u16* qp = Q + (rowBase + q0 + wave * 16 + l16) * H_ + hcol + hs * 8;
#pragma unroll
        for (int c = 0; c < 4; ++c)
            qf[c] = ld_frag2(qp + c * 32, qp + c * 32 + 16);
    }

    v8f o[8] = {};
    float mi[8], li[8];
#pragma unroll
    for (int r = 0; r < 8; ++r) { mi[r] = -3.0e38f; li[r] = 0.0f; }

    const int vrow = tid >> 2, vseg = (tid & 3) * 32;   // 64 rows x 4 col-chunks
    const int nkb = (q0 + QB) / KBLK;

    for (int kb = 0; kb < nkb; ++kb) {
        const int key0 = kb * KBLK;
        // ---- stage K (TDM DMA or manual) and V (manual transpose) ----
#if HAVE_TDM
        if (wave == 0)
            tdm_load_2d((u32)(uintptr_t)&sK[0],
                        Kk + (rowBase + key0) * H_ + hcol, HD_, KBLK, H_, 5, 3);
#else
        {
            const u16* gk = Kk + (rowBase + key0 + vrow) * H_ + hcol + vseg;
#pragma unroll
            for (int j = 0; j < 4; ++j)
                *(v4u*)(sK + vrow * KSTR + vseg + j * 8) = *(const v4u*)(gk + j * 8);
        }
#endif
        {
            const u16* gv = V + (rowBase + key0 + vrow) * H_ + hcol + vseg;
            v4u vv[4];
#pragma unroll
            for (int j = 0; j < 4; ++j) vv[j] = *(const v4u*)(gv + j * 8);
#pragma unroll
            for (int j = 0; j < 32; ++j) {
                u32 w = vv[j >> 3][(j >> 1) & 3];
                sV[(vseg + j) * VSTR + vrow] = (u16)(w >> ((j & 1) * 16));
            }
        }
#if HAVE_TDM
        if (wave == 0) __builtin_amdgcn_s_wait_tensorcnt(0);
#endif
        __syncthreads();

        // ---- S = Q K^T : 16x64 tile (4 C-frags, 16 WMMAs) ----
        v8f st[4] = {};
#pragma unroll
        for (int c = 0; c < 4; ++c) {
            const u16* pk = sK + l16 * KSTR + c * 32 + hs * 8;
#pragma unroll
            for (int nt = 0; nt < 4; ++nt)
                st[nt] = wmma_bf16(qf[c], ld_frag2(pk + nt * 16 * KSTR, pk + nt * 16 * KSTR + 16), st[nt]);
        }

        // ---- scale, causal mask, online softmax ----
        const bool needMask = (key0 + KBLK - 1) > q0;
#pragma unroll
        for (int r = 0; r < 8; ++r) {
            float e[4];
            int qr = q0 + wave * 16 + r + 8 * hs;
#pragma unroll
            for (int nt = 0; nt < 4; ++nt) {
                e[nt] = st[nt][r] * scale;
                if (needMask && (key0 + nt * 16 + l16 > qr)) e[nt] = -3.0e38f;
            }
            float v = fmaxf(fmaxf(e[0], e[1]), fmaxf(e[2], e[3]));
            v = fmaxf(v, __shfl_xor(v, 1, 32));
            v = fmaxf(v, __shfl_xor(v, 2, 32));
            v = fmaxf(v, __shfl_xor(v, 4, 32));
            v = fmaxf(v, __shfl_xor(v, 8, 32));
            float mo = mi[r];
            float mn = fmaxf(mo, v);
            mi[r] = mn;
            float alpha = __expf(mo - mn);
            li[r] *= alpha;
#pragma unroll
            for (int f = 0; f < 8; ++f) o[f][r] *= alpha;
            float rs = 0.0f;
#pragma unroll
            for (int nt = 0; nt < 4; ++nt) {
                float pv = __expf(e[nt] - mn);
                st[nt][r] = pv;
                rs += pv;
            }
            rs += __shfl_xor(rs, 1, 32);
            rs += __shfl_xor(rs, 2, 32);
            rs += __shfl_xor(rs, 4, 32);
            rs += __shfl_xor(rs, 8, 32);
            li[r] += rs;
        }

        // ---- P -> wave-private LDS (C-layout scatter), reload as A-frags ----
        {
            u16* pp = sP + wave * 16 * PSTR;
#pragma unroll
            for (int r = 0; r < 8; ++r) {
                int m = r + 8 * hs;
#pragma unroll
                for (int nt = 0; nt < 4; ++nt)
                    pp[m * PSTR + nt * 16 + l16] = f2bf(st[nt][r]);
            }
        }
        asm volatile("s_wait_dscnt 0" ::: "memory");
        const u16* pf = sP + wave * 16 * PSTR + l16 * PSTR + hs * 8;
        v16bf pa0 = ld_frag2(pf,      pf + 16);
        v16bf pa1 = ld_frag2(pf + 32, pf + 48);

        // ---- O += P V : 2 k-chunks x 8 dv-chunks (16 WMMAs) ----
#pragma unroll
        for (int f = 0; f < 8; ++f) {
            const u16* pv = sV + (f * 16 + l16) * VSTR + hs * 8;
            o[f] = wmma_bf16(pa0, ld_frag2(pv, pv + 16), o[f]);
            o[f] = wmma_bf16(pa1, ld_frag2(pv + 32, pv + 48), o[f]);
        }
        __syncthreads();
    }

    // ---- normalize and write y (bf16, packed (tok, h*128+d)) ----
    {
        float rinv[8];
#pragma unroll
        for (int r = 0; r < 8; ++r) rinv[r] = 1.0f / li[r];
        u16* yp = Y + (rowBase + q0 + wave * 16) * (size_t)H_ + hcol;
#pragma unroll
        for (int f = 0; f < 8; ++f) {
#pragma unroll
            for (int r = 0; r < 8; ++r) {
                int m = r + 8 * hs;
                yp[(size_t)m * H_ + f * 16 + l16] = f2bf(o[f][r] * rinv[r]);
            }
        }
    }
}

// ---------------------------------------------------------------------------
// Host-side orchestration
// ---------------------------------------------------------------------------
extern "C" void kernel_launch(void* const* d_in, const int* in_sizes, int n_in,
                              void* d_out, int out_size, void* d_ws, size_t ws_size,
                              hipStream_t stream) {
    (void)in_sizes; (void)n_in; (void)out_size; (void)ws_size;
    const float* x    = (const float*)d_in[0];
    const float* wkvd = (const float*)d_in[1];
    const float* wqd  = (const float*)d_in[2];
    const float* wku  = (const float*)d_in[3];
    const float* wqu  = (const float*)d_in[4];
    const float* wvu  = (const float*)d_in[5];
    const float* wrk  = (const float*)d_in[6];
    const float* wrq  = (const float*)d_in[7];
    const float* wo   = (const float*)d_in[8];

    u16* p = (u16*)d_ws;
    u16* xb   = p; p += (size_t)MTOK * H_;
    u16* bkvd = p; p += (size_t)LAT_ * H_;
    u16* bqd  = p; p += (size_t)LAT_ * H_;
    u16* bku  = p; p += (size_t)(H_ / 2) * LAT_;
    u16* bqu  = p; p += (size_t)(H_ / 2) * LAT_;
    u16* bvu  = p; p += (size_t)H_ * LAT_;
    u16* brk  = p; p += (size_t)(H_ / 2) * H_;
    u16* brq  = p; p += (size_t)(H_ / 2) * LAT_;
    u16* bwo  = p; p += (size_t)H_ * H_;
    u16* kvd  = p; p += (size_t)MTOK * LAT_;
    u16* qd   = p; p += (size_t)MTOK * LAT_;
    u16* Qb   = p; p += (size_t)MTOK * H_;
    u16* Kb   = p; p += (size_t)MTOK * H_;
    u16* Vb   = p; p += (size_t)MTOK * H_;
    u16* Yb   = p; p += (size_t)MTOK * H_;

    auto cvt = [&](const float* src, u16* dst, size_t n) {
        int g = (int)((n + 1023) / 1024);
        cvt_f32_bf16<<<g, 256, 0, stream>>>(src, dst, (int)n);
    };
    cvt(x,    xb,   (size_t)MTOK * H_);
    cvt(wkvd, bkvd, (size_t)LAT_ * H_);
    cvt(wqd,  bqd,  (size_t)LAT_ * H_);
    cvt(wku,  bku,  (size_t)(H_ / 2) * LAT_);
    cvt(wqu,  bqu,  (size_t)(H_ / 2) * LAT_);
    cvt(wvu,  bvu,  (size_t)H_ * LAT_);
    cvt(wrk,  brk,  (size_t)(H_ / 2) * H_);
    cvt(wrq,  brq,  (size_t)(H_ / 2) * LAT_);
    cvt(wo,   bwo,  (size_t)H_ * H_);

    dim3 blk(256);
    const int GY = MTOK / BM;   // 32

    // kv_d = x @ W_kv_d^T ; q_d = x @ W_q_d^T
    gemm_bf16_wmma<<<dim3(LAT_ / BN, GY), blk, 0, stream>>>(xb, bkvd, kvd, H_,  LAT_, 0, 0, 0);
    gemm_bf16_wmma<<<dim3(LAT_ / BN, GY), blk, 0, stream>>>(xb, bqd,  qd,  H_,  LAT_, 0, 0, 0);
    // k_r = x @ W_rope_k^T  -> K buffer rope slice (head remap, +64)
    gemm_bf16_wmma<<<dim3((H_ / 2) / BN, GY), blk, 0, stream>>>(xb, brk, Kb, H_,  H_, 1, RD_, 0);
    // q_c, q_r from q_d ; k_c, v from kv_d
    gemm_bf16_wmma<<<dim3((H_ / 2) / BN, GY), blk, 0, stream>>>(qd,  bqu, Qb, LAT_, H_, 1, 0,   0);
    gemm_bf16_wmma<<<dim3((H_ / 2) / BN, GY), blk, 0, stream>>>(qd,  brq, Qb, LAT_, H_, 1, RD_, 0);
    gemm_bf16_wmma<<<dim3((H_ / 2) / BN, GY), blk, 0, stream>>>(kvd, bku, Kb, LAT_, H_, 1, 0,   0);
    gemm_bf16_wmma<<<dim3(H_ / BN, GY),       blk, 0, stream>>>(kvd, bvu, Vb, LAT_, H_, 0, 0,   0);

    // RoPE on q_r / k_r slices
    int rg = (B_ * S_ * NH_ * 32) / 256;
    rope_kernel<<<rg, 256, 0, stream>>>(Qb);
    rope_kernel<<<rg, 256, 0, stream>>>(Kb);

    // causal flash attention -> Y
    mla_flash<<<dim3(S_ / QB, NH_, B_), blk, 0, stream>>>(Qb, Kb, Vb, Yb);

    // out = y @ W_o^T (fp32 output)
    gemm_bf16_wmma<<<dim3(H_ / BN, GY), blk, 0, stream>>>(Yb, bwo, (float*)d_out, H_, H_, 0, 0, 1);
}
